// GRUDecoder_18502719111441
// MI455X (gfx1250) — matrix operand order repaired and checked by reading, not compiled, not run on hardware
//
#include <hip/hip_runtime.h>

// =====================================================================
// GRUDecoder forward for MI455X (gfx1250), compile-only target.
//
// All GEMMs run on the WMMA pipe as bf16 16x16x32 with fp32 accumulate
// (8x the MAC density of the fp32 16x16x4 WMMA). fp32 operands are
// converted to bf16 with the native hardware convert while staging
// panels into LDS.
//
// GEMM: full-K LDS panels (K is 128 or 256 everywhere in this model,
// templated + fully unrolled), single barrier per block, 8 waves in a
// 2x4 layout, wave tile 16x32 (two wmma accumulator chains per wave).
// LDS use: 40KB (K=128) / 80KB (K=256) -> >=4 blocks per WGP.
//
// d_in layout assumption (one pointer per pytree leaf, recursive
// insertion order of the setup_inputs() dicts):
//   0 temporal_feature (N,OBS,H)   1 interaction_feature (N,H)   2 goal (N,H)
//   3..6   proj_global: l.w(768,128) l.b ln.g ln.b
//   7..8   conv: wt(30,20) bt(128)
//   9..12  proj_goal:  l.w l.b ln.g ln.b
//   13..20 attn: q.w q.b k.w k.b v.w v.b o.w o.b   (128x128 each)
//   21..26 loc: l1.w l1.b ln.g ln.b l2.w(2,128) l2.b(2)
//   27..62 enc block (36 leaves), 63..98 dec block (36 leaves):
//     +0..3  gru_mid  wih(384,128) whh bih bhh
//     +4..11 mlp_for  l1.w(256,128) l1.b ln1.g ln1.b l2.w(128,256) l2.b ln2.g ln2.b
//     +12..15 gru_for, +16..23 mlp_bac, +24..27 gru_back
//     +28..31 goal_h l.w l.b ln.g ln.b, +32..35 fuse l.w(128,256) l.b ln.g ln.b
//   99 scene_size (==64, hardcoded)
// Workspace requirement: ~830 MB (seven 94MB activation slabs, reused).
// =====================================================================

typedef __bf16 bf16;
typedef __attribute__((ext_vector_type(16))) __bf16 v16bf;
typedef __attribute__((ext_vector_type(8)))  float  v8f;

constexpr int kH = 128, kM = 6, kOBS = 20, kPRED = 30, kN = 1024;
constexpr int kMN = kM * kN;          // 6144
constexpr int kMP = kM * kPRED;       // 180
constexpr int kB  = 16 * kMP;         // 2880 attention batches of 64 tokens

__device__ __forceinline__ bf16 f2bf(float x) { return (bf16)x; }  // native v_cvt
__device__ __forceinline__ float sigm(float x) { return 1.f / (1.f + __expf(-x)); }

// ---------------- WMMA GEMM: C[M,N] = A[M,K] @ W[N,K]^T + bias ----------------
// 256 threads = 8 wave32s arranged 2 (row) x 4 (col).
// Block tile 32 rows x 128 cols; wave tile 16 rows x 32 cols (2 accumulators).
// Full K panel staged in LDS once; K/32 x 2 wmma fully unrolled, barrier-free.
// Requires Mr%32==0, Nc%128==0, bias != null (true for every GEMM here).
template <int K>
__global__ __launch_bounds__(256)
void k_gemm_wmma(const float* __restrict__ A, const float* __restrict__ W,
                 const float* __restrict__ bias, float* __restrict__ C, int Nc)
{
  __shared__ bf16 lA[32 * K];
  __shared__ bf16 lB[128 * K];
  const int tid  = threadIdx.x;
  const int wave = tid >> 5;
  const int lane = tid & 31;
  const int wrow = wave >> 2;           // 0..1
  const int wcol = wave & 3;            // 0..3
  const int rowBase = blockIdx.x * 32;
  const int colBase = blockIdx.y * 128;
  const int rlane = lane & 15;
  const int half  = lane >> 4;

  // ---- stage A panel (32 x K) and B panel (128 x K), float4 -> 4x bf16 ----
  {
    constexpr int tot4 = 32 * K / 4;
#pragma unroll
    for (int i = tid; i < tot4; i += 256) {
      int r = i / (K / 4), k4 = i % (K / 4);
      const float4 v = ((const float4*)(A + (size_t)(rowBase + r) * K))[k4];
      bf16* d = &lA[r * K + k4 * 4];
      d[0] = f2bf(v.x); d[1] = f2bf(v.y); d[2] = f2bf(v.z); d[3] = f2bf(v.w);
    }
  }
  {
    constexpr int tot4 = 128 * K / 4;
#pragma unroll
    for (int i = tid; i < tot4; i += 256) {
      int r = i / (K / 4), k4 = i % (K / 4);
      const float4 v = ((const float4*)(W + (size_t)(colBase + r) * K))[k4];
      bf16* d = &lB[r * K + k4 * 4];
      d[0] = f2bf(v.x); d[1] = f2bf(v.y); d[2] = f2bf(v.z); d[3] = f2bf(v.w);
    }
  }
  __syncthreads();

  v8f acc0 = {0.f, 0.f, 0.f, 0.f, 0.f, 0.f, 0.f, 0.f};
  v8f acc1 = {0.f, 0.f, 0.f, 0.f, 0.f, 0.f, 0.f, 0.f};
  const bf16* aRow  = &lA[(wrow * 16 + rlane) * K];
  const bf16* bRow0 = &lB[(wcol * 32 + rlane) * K];
  const bf16* bRow1 = &lB[(wcol * 32 + 16 + rlane) * K];

  // Fragments (A 16x32, B 32x16) use contiguous k = 16*half + e in both A and B; all runs of
  // 8/16 consecutive bf16 -> compiler emits ds_load_b128 pairs.
#pragma unroll
  for (int k0 = 0; k0 < K; k0 += 32) {
    v16bf fa, fb0, fb1;
#pragma unroll
    for (int e = 0; e < 16; ++e) {
      fa[e]     = aRow[k0 + half * 16 + e];
    }
#pragma unroll
    for (int j = 0; j < 16; ++j) {
      fb0[j] = bRow0[k0 + half * 16 + j];
      fb1[j] = bRow1[k0 + half * 16 + j];
    }
    acc0 = __builtin_amdgcn_wmma_f32_16x16x32_bf16(false, fa, false, fb0,
                                                   (short)0, acc0, false, false);
    acc1 = __builtin_amdgcn_wmma_f32_16x16x32_bf16(false, fa, false, fb1,
                                                   (short)0, acc1, false, false);
  }

  // C layout: lane%16 = N, lane/16 -> M+8, VGPR j = M row.
  const int rbase = rowBase + wrow * 16 + half * 8;
  const int n0 = colBase + wcol * 32 + rlane;
  const int n1 = n0 + 16;
  const float bv0 = bias[n0];
  const float bv1 = bias[n1];
  float* c0 = C + (size_t)rbase * Nc + n0;
  float* c1 = C + (size_t)rbase * Nc + n1;
#pragma unroll
  for (int j = 0; j < 8; ++j) {
    c0[(size_t)j * Nc] = acc0[j] + bv0;
    c1[(size_t)j * Nc] = acc1[j] + bv1;
  }
}

// ---------------- LayerNorm (in-place), mode: 0=none 1=relu 2=sigmoid ----------
__global__ __launch_bounds__(128)
void k_ln(float* __restrict__ x, const float* __restrict__ g,
          const float* __restrict__ b, int D, int mode)
{
  __shared__ float r1[128], r2[128];
  float* xr = x + (size_t)blockIdx.x * D;
  const int tid = threadIdx.x;
  float s = 0.f, s2 = 0.f;
  for (int i = tid; i < D; i += 128) { float v = xr[i]; s += v; s2 += v * v; }
  r1[tid] = s; r2[tid] = s2;
  __syncthreads();
  for (int o = 64; o > 0; o >>= 1) {
    if (tid < o) { r1[tid] += r1[tid + o]; r2[tid] += r2[tid + o]; }
    __syncthreads();
  }
  const float mu = r1[0] / D;
  const float var = r2[0] / D - mu * mu;
  const float rstd = rsqrtf(var + 1e-5f);
  for (int i = tid; i < D; i += 128) {
    float v = (xr[i] - mu) * rstd * g[i] + b[i];
    if (mode == 1) v = fmaxf(v, 0.f);
    else if (mode == 2) v = sigm(v);
    xr[i] = v;
  }
}

// ---------------- elementwise helpers ----------------
__global__ void k_add(const float* __restrict__ a, const float* __restrict__ b,
                      float* __restrict__ c, int n)
{
  int i = blockIdx.x * 256 + threadIdx.x;
  if (i < n) c[i] = a[i] + b[i];
}

__global__ void k_gru(const float* __restrict__ gi, const float* __restrict__ gh,
                      const float* __restrict__ hprev, float* __restrict__ hout, int n)
{
  int i = blockIdx.x * 256 + threadIdx.x;
  if (i >= n) return;
  int row = i >> 7, col = i & 127;
  const float* gir = gi + (size_t)row * 384;
  const float* ghr = gh + (size_t)row * 384;
  float r  = sigm(gir[col] + ghr[col]);
  float z  = sigm(gir[col + 128] + ghr[col + 128]);
  float nn = tanhf(gir[col + 256] + r * ghr[col + 256]);
  hout[i] = (1.f - z) * nn + z * hprev[i];
}

__global__ void k_concat(const float* __restrict__ f, const float* __restrict__ bk,
                         float* __restrict__ o, int n)
{
  int i = blockIdx.x * 256 + threadIdx.x;
  if (i >= n) return;
  int row = i >> 8, c = i & 255;
  o[i] = (c < 128) ? f[(size_t)row * 128 + c] : bk[(size_t)row * 128 + c - 128];
}

__global__ void k_fuse(const float* __restrict__ f, const float* __restrict__ bk,
                       const float* __restrict__ w, float* __restrict__ o, int n)
{
  int i = blockIdx.x * 256 + threadIdx.x;
  if (i < n) o[i] = f[i] * w[i] + bk[i] * (1.f - w[i]);
}

// temporal[p][mn][h] = sum_o wt[p][o]*his[o][mn][h] + bt[h]
__global__ void k_conv(const float* __restrict__ his, const float* __restrict__ wt,
                       const float* __restrict__ bt, float* __restrict__ o, int n)
{
  int i = blockIdx.x * 256 + threadIdx.x;
  if (i >= n) return;
  int h = i & 127, r = i >> 7;
  int mn = r % kMN, p = r / kMN;
  float acc = bt[h];
#pragma unroll
  for (int t = 0; t < kOBS; ++t)
    acc += wt[p * kOBS + t] * his[((size_t)t * kMN + mn) * 128 + h];
  o[i] = acc;
}

// his[t][m*N+n][h] = proj[n*OBS+t][m*128+h]
__global__ void k_his_scatter(const float* __restrict__ proj, float* __restrict__ his, int n)
{
  int i = blockIdx.x * 256 + threadIdx.x;
  if (i >= n) return;
  int h = i & 127, r = i >> 7;
  int an = r % kN, q = r / kN;
  int m = q % kM, t = q / kM;
  his[i] = proj[((size_t)an * kOBS + t) * (kM * kH) + m * kH + h];
}

// goalp[m*N+n][h] = proj[n][m*128+h]
__global__ void k_goal_scatter(const float* __restrict__ proj, float* __restrict__ gp, int n)
{
  int i = blockIdx.x * 256 + threadIdx.x;
  if (i >= n) return;
  int h = i & 127, r = i >> 7;
  int an = r % kN, m = r / kN;
  gp[i] = proj[(size_t)an * (kM * kH) + m * kH + h];
}

__global__ void k_hsoc(const float* __restrict__ itf, float* __restrict__ o, int n)
{
  int i = blockIdx.x * 256 + threadIdx.x;
  if (i >= n) return;
  int h = i & 127, an = (i >> 7) % kN;
  o[i] = itf[(size_t)an * 128 + h];
}

// x[((sc*180+mp)*64+s)][h] = enc[p][m*N + sc*64+s][h],  mp = m*PRED+p
__global__ void k_gatherx(const float* __restrict__ enc, float* __restrict__ x, int n)
{
  int i = blockIdx.x * 256 + threadIdx.x;
  if (i >= n) return;
  int h = i & 127, r = i >> 7;
  int s = r & 63, rr = r >> 6;
  int mp = rr % kMP, sc = rr / kMP;
  int m = mp / kPRED, p = mp % kPRED;
  int an = sc * 64 + s;
  x[i] = enc[((size_t)p * kMN + m * kN + an) * 128 + h];
}

// ref[p][m*N+n][h] = o[((sc*180 + m*PRED+p)*64 + s)][h]
__global__ void k_scatterref(const float* __restrict__ o, float* __restrict__ ref, int n)
{
  int i = blockIdx.x * 256 + threadIdx.x;
  if (i >= n) return;
  int h = i & 127, r = i >> 7;
  int mn = r % kMN, p = r / kMN;
  int m = mn / kN, an = mn % kN;
  int sc = an >> 6, s = an & 63;
  size_t row = ((size_t)sc * kMP + m * kPRED + p) * 64 + s;
  ref[i] = o[row * 128 + h];
}

// per (batch,head) scaled-dot attention over 64 tokens, HD=16
__global__ __launch_bounds__(64)
void k_sdpa(const float* __restrict__ Q, const float* __restrict__ K,
            const float* __restrict__ V, float* __restrict__ O)
{
  __shared__ float sq[64][16], sk[64][16], sv[64][16];
  const int b = blockIdx.x >> 3, hd = blockIdx.x & 7;
  const int t = threadIdx.x;
  const size_t base = (size_t)b * 64 * 128 + hd * 16;
#pragma unroll
  for (int i = 0; i < 16; ++i) {
    sq[t][i] = Q[base + (size_t)t * 128 + i];
    sk[t][i] = K[base + (size_t)t * 128 + i];
    sv[t][i] = V[base + (size_t)t * 128 + i];
  }
  __syncthreads();
  float sc[64], mx = -3.0e38f;
  for (int j = 0; j < 64; ++j) {
    float s = 0.f;
#pragma unroll
    for (int i = 0; i < 16; ++i) s += sq[t][i] * sk[j][i];
    s *= 0.25f;                 // 1/sqrt(HD)
    sc[j] = s;
    mx = fmaxf(mx, s);
  }
  float sum = 0.f, o[16];
#pragma unroll
  for (int i = 0; i < 16; ++i) o[i] = 0.f;
  for (int j = 0; j < 64; ++j) {
    float e = __expf(sc[j] - mx);
    sum += e;
#pragma unroll
    for (int i = 0; i < 16; ++i) o[i] += e * sv[j][i];
  }
  float inv = 1.f / sum;
#pragma unroll
  for (int i = 0; i < 16; ++i) O[base + (size_t)t * 128 + i] = o[i] * inv;
}

// final 128->2 projection, out[(mn*PRED+t)*2+c]
__global__ void k_locl2(const float* __restrict__ x, const float* __restrict__ w2,
                        const float* __restrict__ b2, float* __restrict__ out, int t)
{
  int i = blockIdx.x * 256 + threadIdx.x;
  if (i >= kMN * 2) return;
  int mn = i >> 1, c = i & 1;
  const float* xr = x + (size_t)mn * 128;
  const float* wr = w2 + c * 128;
  float acc = b2[c];
  for (int j = 0; j < 128; ++j) acc += xr[j] * wr[j];
  out[((size_t)mn * kPRED + t) * 2 + c] = acc;
}

// =====================================================================
extern "C" void kernel_launch(void* const* d_in, const int* in_sizes, int n_in,
                              void* d_out, int out_size, void* d_ws, size_t ws_size,
                              hipStream_t stream)
{
  (void)in_sizes; (void)n_in; (void)out_size; (void)ws_size;
  const float* tf  = (const float*)d_in[0];
  const float* itf = (const float*)d_in[1];
  const float* gl  = (const float*)d_in[2];
#define PRM(i) ((const float*)d_in[(i)])

  float* ws = (float*)d_ws;
  size_t off = 0;
  auto alloc = [&](size_t n) { float* p = ws + off; off += n; return p; };
  const size_t SLAB = (size_t)kPRED * kMN * kH;       // 23,592,960 floats
  float* his   = alloc((size_t)kOBS * kMN * kH);
  float* tempo = alloc(SLAB);
  float* hsoc  = alloc((size_t)kMN * kH);
  float* goalp = alloc((size_t)kMN * kH);
  float* hb0   = alloc((size_t)kMN * kH);
  float* gi    = alloc((size_t)kMN * 384);
  float* gh    = alloc((size_t)kMN * 384);
  float* t256  = alloc((size_t)kMN * 256);
  float* cat   = alloc((size_t)kMN * 256);
  float* t128  = alloc((size_t)kMN * kH);
  float* vt    = alloc((size_t)kMN * kH);
  float* S0 = alloc(SLAB); float* S1 = alloc(SLAB); float* S2 = alloc(SLAB);
  float* S3 = alloc(SLAB); float* S4 = alloc(SLAB); float* S5 = alloc(SLAB);
  float* S6 = alloc(SLAB);

  auto gemm = [&](const float* A, const float* W, const float* b, float* C,
                  int Mr, int Nc, int K) {
    dim3 g(Mr / 32, Nc / 128);
    if (K == 128)
      k_gemm_wmma<128><<<g, 256, 0, stream>>>(A, W, b, C, Nc);
    else
      k_gemm_wmma<256><<<g, 256, 0, stream>>>(A, W, b, C, Nc);
  };
  auto lnorm = [&](float* x, const float* g, const float* b, int rows, int D, int mode) {
    k_ln<<<rows, 128, 0, stream>>>(x, g, b, D, mode);
  };
  auto blk = [](int n) { return (n + 255) / 256; };
  const int NH = kMN * kH;

  auto run_block = [&](int base, const float* ref, float* vmid, float* outf,
                       float* outb, float* bout) {
    // hb0 = relu(LN(goal_p @ Wgh^T + b))
    gemm(goalp, PRM(base + 28), PRM(base + 29), hb0, kMN, kH, kH);
    lnorm(hb0, PRM(base + 30), PRM(base + 31), kMN, kH, 1);
    // mid scan: c = GRU(gru_mid, temporal[t], c + ref[t])
    for (int t = 0; t < kPRED; ++t) {
      const float* prev = (t == 0) ? hsoc : (vmid + (size_t)(t - 1) * NH);
      const float* hin = prev;
      if (ref) {
        k_add<<<blk(NH), 256, 0, stream>>>(prev, ref + (size_t)t * NH, vt, NH);
        hin = vt;
      }
      gemm(tempo + (size_t)t * NH, PRM(base + 0), PRM(base + 2), gi, kMN, 384, kH);
      gemm(hin, PRM(base + 1), PRM(base + 3), gh, kMN, 384, kH);
      k_gru<<<blk(NH), 256, 0, stream>>>(gi, gh, hin, vmid + (size_t)t * NH, NH);
    }
    const float* hlast = his + (size_t)(kOBS - 1) * NH;
    // forward: v_for[t] = hsoc + mlp_for(v_mid[t]); c = GRU(gru_for, v_for[t], c)
    for (int t = 0; t < kPRED; ++t) {
      const float* vm = vmid + (size_t)t * NH;
      gemm(vm, PRM(base + 4), PRM(base + 5), t256, kMN, 256, kH);
      lnorm(t256, PRM(base + 6), PRM(base + 7), kMN, 256, 0);
      gemm(t256, PRM(base + 8), PRM(base + 9), t128, kMN, kH, 256);
      lnorm(t128, PRM(base + 10), PRM(base + 11), kMN, kH, 1);
      k_add<<<blk(NH), 256, 0, stream>>>(t128, hsoc, vt, NH);
      const float* prev = (t == 0) ? hlast : (outf + (size_t)(t - 1) * NH);
      gemm(vt, PRM(base + 12), PRM(base + 14), gi, kMN, 384, kH);
      gemm(prev, PRM(base + 13), PRM(base + 15), gh, kMN, 384, kH);
      k_gru<<<blk(NH), 256, 0, stream>>>(gi, gh, prev, outf + (size_t)t * NH, NH);
    }
    // backward (reverse scan from hb0)
    for (int t = kPRED - 1; t >= 0; --t) {
      const float* vm = vmid + (size_t)t * NH;
      gemm(vm, PRM(base + 16), PRM(base + 17), t256, kMN, 256, kH);
      lnorm(t256, PRM(base + 18), PRM(base + 19), kMN, 256, 0);
      gemm(t256, PRM(base + 20), PRM(base + 21), t128, kMN, kH, 256);
      lnorm(t128, PRM(base + 22), PRM(base + 23), kMN, kH, 1);
      k_add<<<blk(NH), 256, 0, stream>>>(t128, hsoc, vt, NH);
      const float* prev = (t == kPRED - 1) ? hb0 : (outb + (size_t)(t + 1) * NH);
      gemm(vt, PRM(base + 24), PRM(base + 26), gi, kMN, 384, kH);
      gemm(prev, PRM(base + 25), PRM(base + 27), gh, kMN, 384, kH);
      k_gru<<<blk(NH), 256, 0, stream>>>(gi, gh, prev, outb + (size_t)t * NH, NH);
    }
    // fuse: w = sigmoid(LN([f,b]@Wfuse^T)); out = f*w + b*(1-w)
    for (int t = 0; t < kPRED; ++t) {
      const float* f  = outf + (size_t)t * NH;
      const float* bk = outb + (size_t)t * NH;
      k_concat<<<blk(kMN * 256), 256, 0, stream>>>(f, bk, cat, kMN * 256);
      gemm(cat, PRM(base + 32), PRM(base + 33), t128, kMN, kH, 256);
      lnorm(t128, PRM(base + 34), PRM(base + 35), kMN, kH, 2);
      k_fuse<<<blk(NH), 256, 0, stream>>>(f, bk, t128, bout + (size_t)t * NH, NH);
    }
  };

  // ---- preprocessing ----
  gemm(tf, PRM(3), PRM(4), S0, kN * kOBS, kM * kH, kH);          // proj_global
  lnorm(S0, PRM(5), PRM(6), kN * kOBS, kM * kH, 1);
  k_his_scatter<<<blk(kOBS * NH), 256, 0, stream>>>(S0, his, kOBS * NH);
  k_conv<<<blk(kPRED * NH), 256, 0, stream>>>(his, PRM(7), PRM(8), tempo, kPRED * NH);
  k_hsoc<<<blk(NH), 256, 0, stream>>>(itf, hsoc, NH);
  gemm(gl, PRM(9), PRM(10), S1, kN, kM * kH, kH);                // proj_goal
  lnorm(S1, PRM(11), PRM(12), kN, kM * kH, 1);
  k_goal_scatter<<<blk(NH), 256, 0, stream>>>(S1, goalp, NH);

  // ---- encoder block (refinement = 0) ----
  run_block(27, nullptr, S0, S1, S2, S5);

  // ---- per-scene self-attention refinement ----
  const int xTot = kB * 64 * kH;                                 // == SLAB
  k_gatherx<<<blk(xTot), 256, 0, stream>>>(S5, S0, xTot);
  gemm(S0, PRM(13), PRM(14), S1, kB * 64, kH, kH);               // q
  gemm(S0, PRM(15), PRM(16), S2, kB * 64, kH, kH);               // k
  gemm(S0, PRM(17), PRM(18), S3, kB * 64, kH, kH);               // v
  k_sdpa<<<kB * 8, 64, 0, stream>>>(S1, S2, S3, S4);             // ctx
  gemm(S4, PRM(19), PRM(20), S0, kB * 64, kH, kH);               // out proj
  k_scatterref<<<blk(kPRED * NH), 256, 0, stream>>>(S0, S6, kPRED * NH);

  // ---- decoder block (refinement = ref) ----
  run_block(63, S6, S0, S1, S2, S5);

  // ---- final location head ----
  float* outp = (float*)d_out;
  for (int t = 0; t < kPRED; ++t) {
    gemm(S5 + (size_t)t * NH, PRM(21), PRM(22), t128, kMN, kH, kH);
    lnorm(t128, PRM(23), PRM(24), kMN, kH, 1);
    k_locl2<<<blk(kMN * 2), 256, 0, stream>>>(t128, PRM(25), PRM(26), outp, t);
  }
#undef PRM
}
